// SinkAwareAttention_21749714387217
// MI455X (gfx1250) — matrix operand order repaired and checked by reading, not compiled, not run on hardware
//
#include <hip/hip_runtime.h>
#include <hip/hip_bf16.h>

#define DEV __device__ __forceinline__

typedef __attribute__((ext_vector_type(16))) _Float16 v16h;
typedef __attribute__((ext_vector_type(8)))  _Float16 v8h;
typedef __attribute__((ext_vector_type(4)))  _Float16 v4h;
typedef __attribute__((ext_vector_type(8)))  float    v8f;
typedef __attribute__((ext_vector_type(4)))  float    v4f;
typedef __attribute__((ext_vector_type(4)))  unsigned int v4u;
typedef __attribute__((ext_vector_type(8)))  int      v8i;
typedef __attribute__((ext_vector_type(4)))  int      v4i;

constexpr int B_ = 2, S_ = 2048, D_ = 1024, H_ = 16, HD_ = 64;
constexpr int M_ = B_ * S_;              // 4096 tokens
constexpr float SCALE_ = 0.125f;         // 1/sqrt(64)
constexpr float SINK_STRENGTH_ = 0.1f;
constexpr int LDS_LD = 72;               // 64 data halves + 8 pad halves (TDM pad)

DEV v8f wmma_f16(v16h a, v16h b, v8f c) {
  return __builtin_amdgcn_wmma_f32_16x16x32_f16(false, a, false, b, (short)0, c, false, false);
}

// A-matrix fragment (16x32 f16), rows at p (row stride ld, K-contiguous).
DEV v16h load_a_frag(const _Float16* __restrict__ p, int ld, int lane) {
  const int lr = lane & 15, hi = lane >> 4;
  const _Float16* row = p + (size_t)lr * ld;
  v8h x0 = *(const v8h*)(row + hi * 8);
  v8h x1 = *(const v8h*)(row + 16 + hi * 8);
  v16h a;
#pragma unroll
  for (int i = 0; i < 8; ++i) { a[i] = x0[i]; a[8 + i] = x1[i]; }
  return a;
}

// B-matrix fragment (32x16 f16) from a row-major [N,K] source at p (NT GEMM).
DEV v16h load_b_frag(const _Float16* __restrict__ p, int ld, int lane) {
  const int lr = lane & 15, hi = lane >> 4;
  const _Float16* row = p + (size_t)lr * ld + hi * 8;
  v8h x0 = *(const v8h*)(row);
  v8h x1 = *(const v8h*)(row + 16);
  v16h b;
#pragma unroll
  for (int i = 0; i < 8; ++i) { b[i] = x0[i]; b[8 + i] = x1[i]; }
  return b;
}

DEV float lane_bcast(float v, int srclane) {  // ds_bpermute broadcast (wave32)
  return __int_as_float(__builtin_amdgcn_ds_bpermute(srclane << 2, __float_as_int(v)));
}

// ---------------------------------------------------------- TDM 2D tile load
// Moves a tile_d1 x tile_d0 (rows x contiguous elems, 2B each) tile from
// global to LDS, padding 4 DWORDs after every 32 DWORDs stored so the LDS
// row stride becomes 72 halves (bank-conflict-free ds_load_b128 afterwards).
DEV void tdm_load_2d(unsigned lds_off, const void* gaddr,
                     unsigned tile_d0, unsigned tile_d1,
                     unsigned tensor_d0, unsigned tensor_d1,
                     unsigned stride0_elems) {
  const unsigned long long ga = (unsigned long long)(size_t)gaddr;
  v4u g0;
  g0[0] = 1u;                                    // count=1
  g0[1] = lds_off;                               // LDS byte address
  g0[2] = (unsigned)(ga & 0xffffffffu);
  g0[3] = (unsigned)((ga >> 32) & 0x1ffffffu) | (2u << 30);  // type=2 (image)
  v8i g1;
  g1[0] = (int)((1u << 16)        // data_size = 1 -> 2 bytes
              | (1u << 20)        // pad_enable
              | (4u << 22)        // pad_interval: after 32 DWORDs
              | (3u << 25));      // pad_amount: 4 DWORDs
  g1[1] = (int)((tensor_d0 & 0xffffu) << 16);
  g1[2] = (int)((tensor_d0 >> 16) | ((tensor_d1 & 0xffffu) << 16));
  g1[3] = (int)(((tensor_d1 >> 16) & 0xffffu) | ((tile_d0 & 0xffffu) << 16));
  g1[4] = (int)(tile_d1 & 0xffffu);              // tile_dim1 (tile_dim2 = 0)
  g1[5] = (int)stride0_elems;                    // tensor_dim0_stride lo32
  g1[6] = 0;
  g1[7] = 0;
  v4i g2 = {0, 0, 0, 0};
  v4i g3 = {0, 0, 0, 0};
#if defined(__clang_major__) && (__clang_major__ >= 23)
  v8i g4 = {0, 0, 0, 0, 0, 0, 0, 0};
  __builtin_amdgcn_tensor_load_to_lds(g0, g1, g2, g3, g4, 0);
#else
  __builtin_amdgcn_tensor_load_to_lds(g0, g1, g2, g3, 0);
#endif
}

// ---------------------------------------------------------------- fp32 -> f16
__global__ __launch_bounds__(256)
void cvt_f32_to_f16(const float* __restrict__ in, _Float16* __restrict__ out, int n) {
  int i = (blockIdx.x * 256 + threadIdx.x) * 4;
  if (i + 3 < n) {
    v4f x = *(const v4f*)(in + i);
    v4h y;
#pragma unroll
    for (int j = 0; j < 4; ++j) y[j] = (_Float16)x[j];
    *(v4h*)(out + i) = y;
  } else {
    for (int j = i; j < n; ++j) out[j] = (_Float16)in[j];
  }
}

// ------------------------------------------------- sink argmin + key bias vec
__global__ __launch_bounds__(256)
void sink_bias_kernel(const float* __restrict__ positions, const int* __restrict__ mask,
                      float* __restrict__ biasvec, int* __restrict__ sink_out) {
  __shared__ float sm[256];
  __shared__ int   si[256];
  const int b = blockIdx.x, tid = threadIdx.x;
  const float INF = __int_as_float(0x7f800000);
  float best = INF; int bidx = 0;
  for (int s = tid; s < S_; s += 256) {
    float p = (mask[b * S_ + s] != 0) ? positions[b * S_ + s] : INF;
    if (p < best) { best = p; bidx = s; }
  }
  sm[tid] = best; si[tid] = bidx;
  __syncthreads();
  for (int off = 128; off > 0; off >>= 1) {
    if (tid < off) {
      if (sm[tid + off] < sm[tid] ||
          (sm[tid + off] == sm[tid] && si[tid + off] < si[tid])) {
        sm[tid] = sm[tid + off]; si[tid] = si[tid + off];
      }
    }
    __syncthreads();
  }
  const int sink = si[0];
  for (int s = tid; s < S_; s += 256) {
    biasvec[b * S_ + s] =
        (mask[b * S_ + s] == 0) ? -1e9f : ((s == sink) ? SINK_STRENGTH_ : 0.f);
  }
  if (tid == 0) sink_out[b] = sink;
}

// ------------------------------------------------------ projection GEMM (NT)
// out[m,n] = sum_k X[m,k]*W[n,k] + bias[n].  64x64 per wave (16 WMMA / 8KB).
// MODE 0: f16 row-major | MODE 1: f16 per-head transposed Vt[b,h,hd,S] | MODE 2: f32
template <int MODE>
__global__ __launch_bounds__(256)
void proj_gemm(const _Float16* __restrict__ X, const _Float16* __restrict__ W,
               const float* __restrict__ bias, void* __restrict__ outp) {
  constexpr int K_ = D_, N_ = D_;
  const int lane = threadIdx.x & 31;
  const int wave = threadIdx.x >> 5;                   // 8 waves: 2(M) x 4(N)
  const int lr = lane & 15, hi = lane >> 4;
  const int m0w = blockIdx.x * 128 + (wave >> 2) * 64;
  const int n0w = blockIdx.y * 256 + (wave & 3) * 64;

  v8f acc[4][4];
#pragma unroll
  for (int mi = 0; mi < 4; ++mi)
#pragma unroll
    for (int ni = 0; ni < 4; ++ni)
#pragma unroll
      for (int r = 0; r < 8; ++r) acc[mi][ni][r] = 0.f;

  for (int k0 = 0; k0 < K_; k0 += 32) {
    v16h a[4], bf[4];
#pragma unroll
    for (int i = 0; i < 4; ++i) {
      a[i]  = load_a_frag(X + (size_t)(m0w + i * 16) * K_ + k0, K_, lane);
      bf[i] = load_b_frag(W + (size_t)(n0w + i * 16) * K_ + k0, K_, lane);
    }
#pragma unroll
    for (int mi = 0; mi < 4; ++mi)
#pragma unroll
      for (int ni = 0; ni < 4; ++ni)
        acc[mi][ni] = wmma_f16(a[mi], bf[ni], acc[mi][ni]);
  }

#pragma unroll
  for (int ni = 0; ni < 4; ++ni) {
    const int n = n0w + ni * 16 + lr;
    const float bn = bias[n];
#pragma unroll
    for (int mi = 0; mi < 4; ++mi) {
#pragma unroll
      for (int r = 0; r < 8; ++r) {
        const int m = m0w + mi * 16 + hi * 8 + r;
        const float v = acc[mi][ni][r] + bn;
        if (MODE == 0) {
          ((_Float16*)outp)[(size_t)m * N_ + n] = (_Float16)v;
        } else if (MODE == 1) {
          const int hh = n >> 6, dh = n & 63;
          const int bb = m / S_, ss = m - bb * S_;
          ((_Float16*)outp)[(((size_t)bb * H_ + hh) * HD_ + dh) * S_ + ss] = (_Float16)v;
        } else {
          ((float*)outp)[(size_t)m * N_ + n] = v;
        }
      }
    }
  }
}

// ----------------------------------------------------- flash attention (f16)
// 4 waves/WG, 32 queries/wave (128 q/WG). K/V tiles (64 keys) staged in LDS
// by the Tensor Data Mover (wave 0, double-buffered, TDM row padding), then
// consumed via ds_load fragments shared across both query sub-tiles.
// Scores are computed TRANSPOSED (K as A, Q as B) so softmaxed P is already
// in A-matrix layout for the P*V WMMA.
__global__ __launch_bounds__(128)
void attn_kernel(const _Float16* __restrict__ Qh, const _Float16* __restrict__ Kh,
                 const _Float16* __restrict__ Vt, const float* __restrict__ biasvec,
                 _Float16* __restrict__ Oout) {
  __shared__ _Float16 lds_k[2][64 * LDS_LD];   // [key][d] rows, stride 72
  __shared__ _Float16 lds_v[2][64 * LDS_LD];   // [d][key] rows, stride 72

  const int lane = threadIdx.x & 31;
  const int wave = threadIdx.x >> 5;           // 0..3
  const int lr = lane & 15, hi = lane >> 4;
  const int bh = blockIdx.y;                   // b*H + h
  const int b = bh >> 4, h = bh & 15;
  const int q0w = blockIdx.x * 128 + wave * 32;

  const _Float16* Qbase = Qh + (size_t)b * S_ * D_ + h * HD_;
  const _Float16* Kbase = Kh + (size_t)b * S_ * D_ + h * HD_;
  const _Float16* Vbase = Vt + (size_t)bh * HD_ * S_;    // [hd][S]
  const float* bias = biasvec + b * S_;

  // Q B-operand fragments: 2 query sub-tiles x 2 d-chunks, kernel-resident
  v16h qb[2][2];
#pragma unroll
  for (int qt = 0; qt < 2; ++qt) {
    qb[qt][0] = load_b_frag(Qbase + (size_t)(q0w + qt * 16) * D_,      D_, lane);
    qb[qt][1] = load_b_frag(Qbase + (size_t)(q0w + qt * 16) * D_ + 32, D_, lane);
  }

  v8f o[2][4];
#pragma unroll
  for (int qt = 0; qt < 2; ++qt)
#pragma unroll
    for (int t = 0; t < 4; ++t)
#pragma unroll
      for (int r = 0; r < 8; ++r) o[qt][t][r] = 0.f;

  float m_run[2] = {-1e30f, -1e30f}, l_run[2] = {0.f, 0.f};

  // prologue: DMA buffer 0 (wave 0 only; TENSORcnt is per-wave)
  if (wave == 0) {
    tdm_load_2d((unsigned)(size_t)&lds_k[0][0], Kbase, 64, 64, D_, S_, D_);
    tdm_load_2d((unsigned)(size_t)&lds_v[0][0], Vbase, 64, 64, S_, HD_, S_);
  }

  constexpr int NIT = S_ / 64;
  for (int it = 0; it < NIT; ++it) {
    const int k0 = it * 64;
    const int buf = it & 1;
    if (wave == 0) __builtin_amdgcn_s_wait_tensorcnt(0);
    __syncthreads();                       // buf ready; prev compute finished
    if (wave == 0 && (it + 1) < NIT) {     // prefetch next tile into buf^1
      tdm_load_2d((unsigned)(size_t)&lds_k[buf ^ 1][0],
                  Kbase + (size_t)(k0 + 64) * D_, 64, 64, D_, S_, D_);
      tdm_load_2d((unsigned)(size_t)&lds_v[buf ^ 1][0],
                  Vbase + (k0 + 64), 64, 64, S_, HD_, S_);
    }
    const _Float16* kt = &lds_k[buf][0];   // [key][d], ld = 72
    const _Float16* vt = &lds_v[buf][0];   // [d][key], ld = 72

    // K / V fragments from LDS, shared by both query sub-tiles
    v16h ka[4][2], vb[4][2];
#pragma unroll
    for (int j = 0; j < 4; ++j) {
      ka[j][0] = load_a_frag(kt + (j * 16) * LDS_LD,      LDS_LD, lane);
      ka[j][1] = load_a_frag(kt + (j * 16) * LDS_LD + 32, LDS_LD, lane);
      vb[j][0] = load_b_frag(vt + (j * 16) * LDS_LD,      LDS_LD, lane);
      vb[j][1] = load_b_frag(vt + (j * 16) * LDS_LD + 32, LDS_LD, lane);
    }
    // bias values: element (j,r) is key = k0 + j*16 + hi*8 + r (same for both qt)
    float bvals[32];
#pragma unroll
    for (int j = 0; j < 4; ++j) {
      const float* bp = bias + k0 + j * 16 + hi * 8;
      v4f b0 = *(const v4f*)(bp);
      v4f b1 = *(const v4f*)(bp + 4);
#pragma unroll
      for (int r = 0; r < 8; ++r) bvals[j * 8 + r] = (r < 4) ? b0[r] : b1[r - 4];
    }

#pragma unroll
    for (int qt = 0; qt < 2; ++qt) {
      // scores^T tiles [16 keys x 16 queries], K-dim = hd = 64
      float s_vals[32];
#pragma unroll
      for (int j = 0; j < 4; ++j) {
        v8f c;
#pragma unroll
        for (int r = 0; r < 8; ++r) c[r] = 0.f;
        c = wmma_f16(ka[j][0], qb[qt][0], c);
        c = wmma_f16(ka[j][1], qb[qt][1], c);
#pragma unroll
        for (int r = 0; r < 8; ++r) s_vals[j * 8 + r] = c[r] * SCALE_ + bvals[j * 8 + r];
      }
      // online softmax: lane pair L / L^16 hold complementary keys of query q=lane%16
      float mx = s_vals[0];
#pragma unroll
      for (int i = 1; i < 32; ++i) mx = fmaxf(mx, s_vals[i]);
      mx = fmaxf(mx, lane_bcast(mx, lane ^ 16));
      const float m_new = fmaxf(m_run[qt], mx);
      const float fscale = __expf(m_run[qt] - m_new);
      float lsum = 0.f;
      _Float16 p16[32];
#pragma unroll
      for (int i = 0; i < 32; ++i) {
        const float e = __expf(s_vals[i] - m_new);
        lsum += e;
        p16[i] = (_Float16)e;
      }
      lsum += lane_bcast(lsum, lane ^ 16);
      l_run[qt] = l_run[qt] * fscale + lsum;
      m_run[qt] = m_new;
      // rescale O rows (row r holds query hi*8 + r of this sub-tile)
      float frow[8];
#pragma unroll
      for (int r = 0; r < 8; ++r) frow[r] = lane_bcast(fscale, hi * 8 + r);
#pragma unroll
      for (int t = 0; t < 4; ++t)
#pragma unroll
        for (int r = 0; r < 8; ++r) o[qt][t][r] *= frow[r];
      // pack P (already in A-layout thanks to transposed scores)
      v16h pa0, pa1;
#pragma unroll
      for (int i = 0; i < 16; ++i) { pa0[i] = p16[i]; pa1[i] = p16[16 + i]; }
      // O += P * V
#pragma unroll
      for (int t = 0; t < 4; ++t) {
        o[qt][t] = wmma_f16(pa0, vb[t][0], o[qt][t]);
        o[qt][t] = wmma_f16(pa1, vb[t][1], o[qt][t]);
      }
    }
  }

  // finalize: divide by l per query row; store f16 [B,S,D]
  _Float16* Obase = Oout + (size_t)b * S_ * D_ + h * HD_;
#pragma unroll
  for (int qt = 0; qt < 2; ++qt) {
    float linv[8];
#pragma unroll
    for (int r = 0; r < 8; ++r) linv[r] = 1.f / lane_bcast(l_run[qt], hi * 8 + r);
#pragma unroll
    for (int t = 0; t < 4; ++t)
#pragma unroll
      for (int r = 0; r < 8; ++r)
        Obase[(size_t)(q0w + qt * 16 + hi * 8 + r) * D_ + t * 16 + lr] =
            (_Float16)(o[qt][t][r] * linv[r]);
  }
}

// ---------------------------------------------------------------------------
extern "C" void kernel_launch(void* const* d_in, const int* in_sizes, int n_in,
                              void* d_out, int out_size, void* d_ws, size_t ws_size,
                              hipStream_t stream) {
  const float* q_in  = (const float*)d_in[0];
  const float* k_in  = (const float*)d_in[1];
  const float* v_in  = (const float*)d_in[2];
  const float* pos   = (const float*)d_in[3];
  const int*   amask = (const int*)d_in[4];
  const float* Wq = (const float*)d_in[5];
  const float* bq = (const float*)d_in[6];
  const float* Wk = (const float*)d_in[7];
  const float* bk = (const float*)d_in[8];
  const float* Wv = (const float*)d_in[9];
  const float* bv = (const float*)d_in[10];
  const float* Wo = (const float*)d_in[11];
  const float* bo = (const float*)d_in[12];

  char* ws = (char*)d_ws;
  const size_t XB = (size_t)M_ * D_ * sizeof(_Float16);  // 8 MiB
  const size_t WB = (size_t)D_ * D_ * sizeof(_Float16);  // 2 MiB
  _Float16* qh  = (_Float16*)(ws + 0 * XB);
  _Float16* kh  = (_Float16*)(ws + 1 * XB);
  _Float16* vt  = (_Float16*)(ws + 2 * XB);              // [B,H,hd,S]
  _Float16* xq  = (_Float16*)(ws + 3 * XB);
  _Float16* xk  = (_Float16*)(ws + 4 * XB);
  _Float16* xv  = (_Float16*)(ws + 5 * XB);
  _Float16* ao  = (_Float16*)(ws + 6 * XB);
  _Float16* whq = (_Float16*)(ws + 7 * XB + 0 * WB);
  _Float16* whk = (_Float16*)(ws + 7 * XB + 1 * WB);
  _Float16* whv = (_Float16*)(ws + 7 * XB + 2 * WB);
  _Float16* who = (_Float16*)(ws + 7 * XB + 3 * WB);
  float* biasvec = (float*)(ws + 7 * XB + 4 * WB);       // [B,S]

  float* outf = (float*)d_out;
  int* sinkp = (int*)(outf + (size_t)M_ * D_);           // tuple tail: sink_indices

  const int NX = M_ * D_;
  const int NW = D_ * D_;
  dim3 cb(256);
  cvt_f32_to_f16<<<dim3((NX / 4 + 255) / 256), cb, 0, stream>>>(q_in, xq, NX);
  cvt_f32_to_f16<<<dim3((NX / 4 + 255) / 256), cb, 0, stream>>>(k_in, xk, NX);
  cvt_f32_to_f16<<<dim3((NX / 4 + 255) / 256), cb, 0, stream>>>(v_in, xv, NX);
  cvt_f32_to_f16<<<dim3((NW / 4 + 255) / 256), cb, 0, stream>>>(Wq, whq, NW);
  cvt_f32_to_f16<<<dim3((NW / 4 + 255) / 256), cb, 0, stream>>>(Wk, whk, NW);
  cvt_f32_to_f16<<<dim3((NW / 4 + 255) / 256), cb, 0, stream>>>(Wv, whv, NW);
  cvt_f32_to_f16<<<dim3((NW / 4 + 255) / 256), cb, 0, stream>>>(Wo, who, NW);

  sink_bias_kernel<<<dim3(B_), cb, 0, stream>>>(pos, amask, biasvec, sinkp);

  dim3 gg(M_ / 128, D_ / 256), gb(256);
  proj_gemm<0><<<gg, gb, 0, stream>>>(xq, whq, bq, qh);
  proj_gemm<0><<<gg, gb, 0, stream>>>(xk, whk, bk, kh);
  proj_gemm<1><<<gg, gb, 0, stream>>>(xv, whv, bv, vt);

  attn_kernel<<<dim3(S_ / 128, B_ * H_), dim3(128), 0, stream>>>(qh, kh, vt, biasvec, ao);

  proj_gemm<2><<<gg, gb, 0, stream>>>(ao, who, bo, outf);
}
